// GroupedQueryAttention_34711925686686
// MI455X (gfx1250) — compile-verified, matrix operand order checked by reading, not run on hardware
//
#include <hip/hip_runtime.h>
#include <hip/hip_bf16.h>

typedef __attribute__((ext_vector_type(16))) _Float16 v16h;
typedef __attribute__((ext_vector_type(8)))  _Float16 v8h;
typedef __attribute__((ext_vector_type(8)))  float    v8f;

union FragU { v16h v; v8h h[2]; };

#define B_SZ   2
#define N_SEQ  2048
#define E_DIM  2048
#define KVD    512
#define DQ     64
#define NQH    32
#define GRP    4
#define M_ROWS (B_SZ * N_SEQ)   // 4096

// ---------------------------------------------------------------------------
// f32 -> f16 conversion (grid-stride, no divisions)
// ---------------------------------------------------------------------------
__global__ void cvt_f32_to_f16(const float* __restrict__ in,
                               _Float16* __restrict__ out, size_t n) {
    size_t i = (size_t)blockIdx.x * blockDim.x + threadIdx.x;
    size_t stride = (size_t)gridDim.x * blockDim.x;
    for (; i < n; i += stride) out[i] = (_Float16)in[i];
}

// f32 (K x N) -> f16 transposed (N x K). grid = (K/256, N); k from tid.
__global__ void cvt_transpose_f32_to_f16(const float* __restrict__ in,
                                         _Float16* __restrict__ out,
                                         int K, int N) {
    int k = blockIdx.x * blockDim.x + threadIdx.x;
    int n = blockIdx.y;
    out[(size_t)n * K + k] = (_Float16)in[(size_t)k * N + n];
}

// f16 (B, S, KVD) -> f16 (B, KVD, S). grid = (S/256, KVD, B).
__global__ void transpose_v_f16(const _Float16* __restrict__ in,
                                _Float16* __restrict__ out) {
    int s = blockIdx.x * blockDim.x + threadIdx.x;
    int c = blockIdx.y;
    int b = blockIdx.z;
    out[((size_t)b * KVD + c) * N_SEQ + s] = in[((size_t)b * N_SEQ + s) * KVD + c];
}

// ---------------------------------------------------------------------------
// NT WMMA GEMM: C(M,N) = A(M,K) * Bt(N,K)^T; A,Bt f16 row-major (K-major).
// Block = 256 threads (8 waves, 4x2), tile 128x64, K-step 32.
// All LDS traffic b128; fragments = 2 x ds_load_b128 per ISA f16 layout.
// ---------------------------------------------------------------------------
#define TM 128
#define TN 64
#define TK 32
#define LDK (TK + 16)   // 48 halves = 96B row stride (keeps 16B/32B alignment)

template <bool OUT_F16>
__global__ __launch_bounds__(256)
void gemm_wmma_nt(const _Float16* __restrict__ A,
                  const _Float16* __restrict__ Bt,
                  void* __restrict__ C, int M, int N, int K) {
    __shared__ __align__(32) _Float16 sA[TM][LDK];
    __shared__ __align__(32) _Float16 sB[TN][LDK];

    const int tid  = threadIdx.x;
    const int lane = tid & 31;
    const int wave = tid >> 5;
    const int wm   = wave >> 1;          // 0..3
    const int wn   = wave & 1;           // 0..1
    const int blockM = blockIdx.y * TM;
    const int blockN = blockIdx.x * TN;

    const int lrow = lane & 15;
    const int hi   = lane >> 4;          // 0/1
    const int kb   = hi * 8;             // K-half base per ISA A/B layout

    v8f acc[2][2] = {};

    for (int k0 = 0; k0 < K; k0 += TK) {
        {   // A tile 128x32: one 32B vector per thread
            int r = tid >> 1;
            int c = (tid & 1) * 16;
            const _Float16* src = A + (size_t)(blockM + r) * K + k0 + c;
            *((v16h*)&sA[r][c]) = *((const v16h*)src);
            if (k0 + TK < K)
                __builtin_prefetch(src + TK, 0, 0);     // global_prefetch_b8
        }
        {   // Bt tile 64x32: one 16B vector per thread
            int r = tid >> 2;
            int c = (tid & 3) * 8;
            const _Float16* src = Bt + (size_t)(blockN + r) * K + k0 + c;
            *((v8h*)&sB[r][c]) = *((const v8h*)src);
        }
        __syncthreads();

        v16h af[2], bf[2];
        #pragma unroll
        for (int i = 0; i < 2; ++i) {
            const _Float16* p = &sA[wm * 32 + i * 16 + lrow][0];
            FragU u;
            u.h[0] = *((const v8h*)(p + kb));
            u.h[1] = *((const v8h*)(p + kb + 16));
            af[i] = u.v;
        }
        #pragma unroll
        for (int j = 0; j < 2; ++j) {
            const _Float16* p = &sB[wn * 32 + j * 16 + lrow][0];
            FragU u;
            u.h[0] = *((const v8h*)(p + kb));
            u.h[1] = *((const v8h*)(p + kb + 16));
            bf[j] = u.v;
        }
        #pragma unroll
        for (int i = 0; i < 2; ++i)
            #pragma unroll
            for (int j = 0; j < 2; ++j)
                acc[i][j] = __builtin_amdgcn_wmma_f32_16x16x32_f16(
                    false, af[i], false, bf[j], (short)0, acc[i][j], false, false);
        __syncthreads();
    }

    // Epilogue per ISA C-layout: row = base + r + 8*hi, col = base + (lane&15)
    #pragma unroll
    for (int i = 0; i < 2; ++i) {
        #pragma unroll
        for (int j = 0; j < 2; ++j) {
            int col = blockN + wn * 32 + j * 16 + lrow;
            #pragma unroll
            for (int r = 0; r < 8; ++r) {
                int row = blockM + wm * 32 + i * 16 + r + hi * 8;
                size_t idx = (size_t)row * N + col;
                if (OUT_F16) ((_Float16*)C)[idx] = (_Float16)acc[i][j][r];
                else         ((float*)C)[idx]    = acc[i][j][r];
            }
        }
    }
}

// ---------------------------------------------------------------------------
// Causal flash attention, GQA. One wave per 16-row Q tile of one q-head.
// Key step = 32. Tile-wide running max (exact: common per-row shift),
// row-sums accumulated by a 5th WMMA  P x ones(32x16)  (rides same rescale).
// Q/K/V fragments: 2 x 16B global loads. P restaged via padded LDS.
// ---------------------------------------------------------------------------
__global__ __launch_bounds__(128)
void gqa_flash_wmma(const _Float16* __restrict__ Q,
                    const _Float16* __restrict__ K,
                    const _Float16* __restrict__ Vt,
                    _Float16* __restrict__ O) {
    __shared__ __align__(32) _Float16 sP[4][16][48];   // 96B row stride

    const int lane  = threadIdx.x & 31;
    const int wave  = threadIdx.x >> 5;
    const int qhead = blockIdx.y;
    const int b     = blockIdx.z;
    const int qBase = blockIdx.x * 64 + wave * 16;
    const int lrow  = lane & 15;
    const int hi    = lane >> 4;
    const int kb    = hi * 8;

    const _Float16* Qp = Q  + (size_t)b * N_SEQ * E_DIM + (size_t)qhead * DQ;
    const _Float16* Kp = K  + (size_t)b * N_SEQ * KVD   + (size_t)(qhead / GRP) * DQ;
    const _Float16* Vp = Vt + (size_t)b * KVD * N_SEQ   + (size_t)(qhead / GRP) * DQ * N_SEQ;
    _Float16*       Op = O  + (size_t)b * N_SEQ * E_DIM + (size_t)qhead * DQ;

    // Q fragments (16x64), pre-scaled by 1/sqrt(64) = 0.125 (exact in f16)
    v16h qsv, onesf;
    #pragma unroll
    for (int t = 0; t < 16; ++t) { qsv[t] = (_Float16)0.125f; onesf[t] = (_Float16)1.0f; }

    v16h qf[2];
    #pragma unroll
    for (int c = 0; c < 2; ++c) {
        const _Float16* qp = Qp + (size_t)(qBase + lrow) * E_DIM + c * 32;
        FragU u;
        u.h[0] = *((const v8h*)(qp + kb));
        u.h[1] = *((const v8h*)(qp + kb + 16));
        qf[c] = u.v * qsv;
    }

    float m_run = -1e30f;
    v8f acc[4] = {};
    v8f accl = {};          // running row-sums via P x ones

    for (int s0 = 0; s0 < qBase + 16; s0 += 32) {
        // ---- S = Q * K^T for 32 keys (two 16x16 tiles, D split 0-31/32-63)
        v8f S[2];
        #pragma unroll
        for (int j = 0; j < 2; ++j) {
            const _Float16* kp = Kp + (size_t)(s0 + j * 16 + lrow) * KVD;
            FragU k0, k1;
            k0.h[0] = *((const v8h*)(kp + kb));
            k0.h[1] = *((const v8h*)(kp + kb + 16));
            k1.h[0] = *((const v8h*)(kp + 32 + kb));
            k1.h[1] = *((const v8h*)(kp + 32 + kb + 16));
            v8f z = {};
            z = __builtin_amdgcn_wmma_f32_16x16x32_f16(false, qf[0], false, k0.v,
                                                       (short)0, z, false, false);
            z = __builtin_amdgcn_wmma_f32_16x16x32_f16(false, qf[1], false, k1.v,
                                                       (short)0, z, false, false);
            S[j] = z;
        }

        // ---- causal mask, only on the diagonal block ----
        if (s0 + 31 > qBase) {
            #pragma unroll
            for (int j = 0; j < 2; ++j) {
                int key = s0 + j * 16 + lrow;
                #pragma unroll
                for (int r = 0; r < 8; ++r)
                    if (key > qBase + r + hi * 8) S[j][r] = -1e30f;
            }
        }

        // ---- tile-wide running max (one 5-step wave reduction) ----
        float tmax = S[0][0];
        #pragma unroll
        for (int r = 1; r < 8; ++r) tmax = fmaxf(tmax, S[0][r]);
        #pragma unroll
        for (int r = 0; r < 8; ++r) tmax = fmaxf(tmax, S[1][r]);
        #pragma unroll
        for (int off = 16; off >= 1; off >>= 1)
            tmax = fmaxf(tmax, __shfl_xor(tmax, off, 32));
        float mnew = fmaxf(m_run, tmax);
        float corr = __expf(m_run - mnew);
        m_run = mnew;
        #pragma unroll
        for (int r = 0; r < 8; ++r) {
            accl[r] *= corr;
            #pragma unroll
            for (int c = 0; c < 4; ++c) acc[c][r] *= corr;
        }

        // ---- P = exp(S - m): C-layout -> row-major f16 LDS (scatter) ----
        #pragma unroll
        for (int j = 0; j < 2; ++j)
            #pragma unroll
            for (int r = 0; r < 8; ++r)
                sP[wave][r + hi * 8][j * 16 + lrow] = (_Float16)__expf(S[j][r] - mnew);

        // ---- reload P as A-fragment: 2 x ds_load_b128 ----
        FragU pu;
        {
            const _Float16* p = &sP[wave][lrow][0];
            pu.h[0] = *((const v8h*)(p + kb));
            pu.h[1] = *((const v8h*)(p + kb + 16));
        }

        // ---- row sums: accl += P x ones (WMMA, no lane reduction) ----
        accl = __builtin_amdgcn_wmma_f32_16x16x32_f16(
            false, pu.v, false, onesf, (short)0, accl, false, false);

        // ---- O += P * V  (4 dim-chunks of 16; V key-contiguous via Vt) ----
        #pragma unroll
        for (int c = 0; c < 4; ++c) {
            const _Float16* vp = Vp + (size_t)(c * 16 + lrow) * N_SEQ + s0;
            FragU vu;
            vu.h[0] = *((const v8h*)(vp + kb));
            vu.h[1] = *((const v8h*)(vp + kb + 16));
            acc[c] = __builtin_amdgcn_wmma_f32_16x16x32_f16(
                false, pu.v, false, vu.v, (short)0, acc[c], false, false);
        }
    }

    // ---- normalize + store f16 (feeds Wo GEMM) ----
    #pragma unroll
    for (int c = 0; c < 4; ++c)
        #pragma unroll
        for (int r = 0; r < 8; ++r)
            Op[(size_t)(qBase + r + hi * 8) * E_DIM + c * 16 + lrow] =
                (_Float16)(acc[c][r] / accl[r]);
}

// ---------------------------------------------------------------------------
// Host side
// ---------------------------------------------------------------------------
extern "C" void kernel_launch(void* const* d_in, const int* in_sizes, int n_in,
                              void* d_out, int out_size, void* d_ws, size_t ws_size,
                              hipStream_t stream) {
    const float* x  = (const float*)d_in[0];
    const float* Wq = (const float*)d_in[1];
    const float* Wk = (const float*)d_in[2];
    const float* Wv = (const float*)d_in[3];
    const float* Wo = (const float*)d_in[4];

    char* ws = (char*)d_ws;
    size_t off = 0;
    _Float16* xh   = (_Float16*)(ws + off); off += (size_t)M_ROWS * E_DIM * 2;
    _Float16* wqT  = (_Float16*)(ws + off); off += (size_t)E_DIM * E_DIM * 2;  // (E out) x (E in)
    _Float16* wkT  = (_Float16*)(ws + off); off += (size_t)KVD * E_DIM * 2;    // (KVD) x (E)
    _Float16* wvT  = (_Float16*)(ws + off); off += (size_t)KVD * E_DIM * 2;
    _Float16* woT  = (_Float16*)(ws + off); off += (size_t)E_DIM * E_DIM * 2;
    _Float16* Qh   = (_Float16*)(ws + off); off += (size_t)M_ROWS * E_DIM * 2;
    _Float16* Kh   = (_Float16*)(ws + off); off += (size_t)M_ROWS * KVD * 2;
    _Float16* Vh   = (_Float16*)(ws + off); off += (size_t)M_ROWS * KVD * 2;
    _Float16* VtT  = (_Float16*)(ws + off); off += (size_t)M_ROWS * KVD * 2;   // (B, KVD, N)
    _Float16* Ah   = (_Float16*)(ws + off); off += (size_t)M_ROWS * E_DIM * 2;

    // 1) convert x; convert+transpose weights (one-shot, amortized over GEMMs)
    cvt_f32_to_f16<<<1024, 256, 0, stream>>>(x, xh, (size_t)M_ROWS * E_DIM);
    cvt_transpose_f32_to_f16<<<dim3(E_DIM / 256, E_DIM), 256, 0, stream>>>(Wq, wqT, E_DIM, E_DIM);
    cvt_transpose_f32_to_f16<<<dim3(E_DIM / 256, KVD),   256, 0, stream>>>(Wk, wkT, E_DIM, KVD);
    cvt_transpose_f32_to_f16<<<dim3(E_DIM / 256, KVD),   256, 0, stream>>>(Wv, wvT, E_DIM, KVD);
    cvt_transpose_f32_to_f16<<<dim3(E_DIM / 256, E_DIM), 256, 0, stream>>>(Wo, woT, E_DIM, E_DIM);

    // 2) Q/K/V projections (NT, f16 out)
    gemm_wmma_nt<true><<<dim3(E_DIM / TN, M_ROWS / TM), 256, 0, stream>>>(
        xh, wqT, (void*)Qh, M_ROWS, E_DIM, E_DIM);
    gemm_wmma_nt<true><<<dim3(KVD / TN, M_ROWS / TM), 256, 0, stream>>>(
        xh, wkT, (void*)Kh, M_ROWS, KVD, E_DIM);
    gemm_wmma_nt<true><<<dim3(KVD / TN, M_ROWS / TM), 256, 0, stream>>>(
        xh, wvT, (void*)Vh, M_ROWS, KVD, E_DIM);

    // 3) V -> Vt (key-contiguous layout for P*V B-fragments)
    transpose_v_f16<<<dim3(N_SEQ / 256, KVD, B_SZ), 256, 0, stream>>>(Vh, VtT);

    // 4) causal GQA flash attention (f16 out in (b,n,E) head-concat layout)
    gqa_flash_wmma<<<dim3(N_SEQ / 64, NQH, B_SZ), 128, 0, stream>>>(Qh, Kh, VtT, Ah);

    // 5) output projection (NT, f32 -> d_out)
    gemm_wmma_nt<false><<<dim3(E_DIM / TN, M_ROWS / TM), 256, 0, stream>>>(
        Ah, woT, d_out, M_ROWS, E_DIM, E_DIM);
}
